// RelativeSpatialEncoding_9612136808564
// MI455X (gfx1250) — matrix operand order repaired and checked by reading, compiled, not run on hardware
//
#include <hip/hip_runtime.h>
#include <hip/hip_bf16.h>
#include <type_traits>

typedef __bf16 bf16_t;
typedef __attribute__((ext_vector_type(2)))  __bf16 v2bf;
typedef __attribute__((ext_vector_type(16))) __bf16 v16bf;
typedef __attribute__((ext_vector_type(8)))  float  v8f;

#define C_IN    64
#define C_OUT   64
#define NNODES  50000
#define WAVES_PER_BLOCK 8

union BfFrag {
    uint4 u4[2];   // two 16-byte LDS loads
    v16bf v;       // 16 bf16 = one WMMA operand fragment
};

// ---------------------------------------------------------------------------
// Kernel 1: zero output (max half must start at 0.0f bits), sums and counts.
// ---------------------------------------------------------------------------
__global__ __launch_bounds__(256) void rse_init_kernel(float* __restrict__ out,
                                                       float* __restrict__ sum_ws,
                                                       float* __restrict__ cnt_ws) {
    const long n_out = (long)NNODES * 128;
    const long n_sum = (long)NNODES * C_OUT;
    const long total = n_out + n_sum + NNODES;
    for (long i = (long)blockIdx.x * 256 + threadIdx.x; i < total;
         i += (long)gridDim.x * 256) {
        if (i < n_out)              out[i] = 0.0f;
        else if (i < n_out + n_sum) sum_ws[i - n_out] = 0.0f;
        else                        cnt_ws[i - n_out - n_sum] = 0.0f;
    }
}

// ---------------------------------------------------------------------------
// Kernel 2: per-edge  relu((G[src]-RSC[dst]) @ W^T + b)  via bf16 WMMA,
// scattered into sum/max/count buffers with L2-resident atomics.
// Each wave processes 16 edges per WMMA group (M=16).
// ---------------------------------------------------------------------------
__global__ __launch_bounds__(256) void rse_edge_kernel(
    const float* __restrict__ G, const float* __restrict__ RSC,
    const int*  __restrict__ src, const int* __restrict__ dst,
    const float* __restrict__ W, const float* __restrict__ b,
    float* __restrict__ out, float* __restrict__ sum_ws,
    float* __restrict__ cnt_ws, int E) {

    __shared__ bf16_t sW[C_OUT * C_IN];                 // 8 KB: W in bf16
    __shared__ bf16_t sRel[WAVES_PER_BLOCK][16 * C_IN]; // 16 KB: per-wave A staging
    __shared__ int    sDst[WAVES_PER_BLOCK][16];        // per-wave dst ids

    const int tid  = threadIdx.x;
    const int lane = tid & 31;
    const int wave = tid >> 5;
    const int nlo  = lane & 15;   // column within 16-wide tile / row M of A
    const int half = lane >> 4;   // lane half selects K sub-blocks

    // --- stage W as bf16 in LDS (once per block) ---
    for (int i = tid; i < C_OUT * C_IN; i += 256)
        sW[i] = (bf16_t)W[i];
    __syncthreads();

    // --- B fragments resident in registers: B[k][n] = W[n][k] ---
    // 4 N-tiles x 2 K-halves; lane holds col n=nlo, K-quarter via `half`.
    BfFrag Bf[4][2];
#pragma unroll
    for (int t = 0; t < 4; ++t)
#pragma unroll
        for (int h = 0; h < 2; ++h) {
            const int kbase = h * 32 + half * 16;
            const uint4* p =
                (const uint4*)&sW[(t * 16 + nlo) * C_IN + kbase];
            Bf[t][h].u4[0] = p[0];
            Bf[t][h].u4[1] = p[1];
        }

    float bias[4];
#pragma unroll
    for (int t = 0; t < 4; ++t) bias[t] = b[t * 16 + nlo];

    const int n_groups = (E + 15) >> 4;
    const int gwave    = blockIdx.x * WAVES_PER_BLOCK + wave;
    const int gstride  = gridDim.x * WAVES_PER_BLOCK;

    bf16_t* rel = sRel[wave];

    // Group body, specialized on "all 16 edges valid" (the common case) so
    // the scatter has no per-element exec-mask churn.
    auto do_group = [&](int base, auto full_c) {
        constexpr bool FULL = decltype(full_c)::value;

        // dst ids for this group -> LDS (single guarded store, no per-e cmpx)
        if (lane < 16) {
            int edge = base + lane;
            int ce   = FULL ? edge : (edge < E ? edge : E - 1);
            sDst[wave][lane] = dst[ce];
        }

        // ---- stage rel = G[src] - RSC[dst] for 16 edges into LDS (bf16) ----
#pragma unroll 4
        for (int e = 0; e < 16; ++e) {
            int edge = base + e;
            int ce   = FULL ? edge : (edge < E ? edge : E - 1);
            int s = src[ce];
            int d = dst[ce];
            const float2 gv = *(const float2*)&G[(size_t)s * C_IN + lane * 2];
            const float2 rv = *(const float2*)&RSC[(size_t)d * C_IN + lane * 2];
            v2bf pr;
            pr.x = (bf16_t)(gv.x - rv.x);
            pr.y = (bf16_t)(gv.y - rv.y);
            *(v2bf*)&rel[e * C_IN + lane * 2] = pr;
        }
        // wave-local LDS visibility (single wave owns this region)
        asm volatile("s_wait_dscnt 0" ::: "memory");
        __builtin_amdgcn_wave_barrier();

        // ---- 16x64 @ 64x64 via 8 WMMAs: 4 N-tiles x 2 K-steps ----
        v8f acc[4] = {};
#pragma unroll
        for (int h = 0; h < 2; ++h) {
            // A fragment: lane row m=nlo, K sub-blocks selected by `half`
            BfFrag Af;
            Af.u4[0] = *(const uint4*)&rel[nlo * C_IN + h * 32 + half * 16];
            Af.u4[1] = *(const uint4*)&rel[nlo * C_IN + h * 32 + half * 16 + 8];
#pragma unroll
            for (int t = 0; t < 4; ++t)
                acc[t] = __builtin_amdgcn_wmma_f32_16x16x32_bf16(
                    false, Af.v, false, Bf[t][h].v, (short)0, acc[t],
                    false, false);
        }

        // ---- bias + relu + scatter: r (edge row) outer so the 4 N-tile
        // atomics share one address base and differ by a constant offset ----
#pragma unroll
        for (int r = 0; r < 8; ++r) {
            const int m = r + half * 8;            // edge row in group
            if (!FULL && (base + m) >= E) continue;
            const unsigned d  = (unsigned)sDst[wave][m];
            float* __restrict__ sp = sum_ws + (size_t)d * C_OUT + nlo;
            unsigned* __restrict__ mp =
                (unsigned*)(out + (size_t)d * 128 + nlo);
#pragma unroll
            for (int t = 0; t < 4; ++t) {
                float v = acc[t][r] + bias[t];
                v = v > 0.0f ? v : 0.0f;
                atomicAdd(sp + t * 16, v);
                // enc >= 0, so uint-bit atomicMax == float max; out max half
                // pre-zeroed -> empty nodes stay 0.
                atomicMax(mp + t * 16, __float_as_uint(v));
            }
        }
        // ---- per-edge count ----
        if (lane < 16 && (FULL || (base + lane) < E))
            atomicAdd(&cnt_ws[sDst[wave][lane]], 1.0f);
    };

    for (int g = gwave; g < n_groups; g += gstride) {
        const int base = g << 4;

        // prefetch next chunk of edge indices into GL2
        const int ng = g + gstride;
        if (ng < n_groups) {
            __builtin_prefetch(&src[ng << 4], 0, 0);
            __builtin_prefetch(&dst[ng << 4], 0, 0);
        }

        if (base + 16 <= E) do_group(base, std::true_type{});
        else                do_group(base, std::false_type{});
    }
}

// ---------------------------------------------------------------------------
// Kernel 3: avg = sum / max(cnt,1) into out[:, 64:128]
// ---------------------------------------------------------------------------
__global__ __launch_bounds__(256) void rse_final_kernel(
    const float* __restrict__ sum_ws, const float* __restrict__ cnt_ws,
    float* __restrict__ out) {
    const long i = (long)blockIdx.x * 256 + threadIdx.x;
    if (i >= (long)NNODES * C_OUT) return;
    const long node = i >> 6;
    const int  n    = (int)(i & 63);
    const float c = cnt_ws[node];
    const float denom = c > 1.0f ? c : 1.0f;
    out[node * 128 + 64 + n] = sum_ws[i] / denom;
}

// ---------------------------------------------------------------------------
extern "C" void kernel_launch(void* const* d_in, const int* in_sizes, int n_in,
                              void* d_out, int out_size, void* d_ws,
                              size_t ws_size, hipStream_t stream) {
    const float* G   = (const float*)d_in[0];
    const float* RSC = (const float*)d_in[1];
    const int*   src = (const int*)d_in[2];
    const int*   dst = (const int*)d_in[3];
    const float* W   = (const float*)d_in[4];
    const float* b   = (const float*)d_in[5];
    float* out = (float*)d_out;
    const int E = in_sizes[2];

    float* sum_ws = (float*)d_ws;                     // NNODES * 64 floats
    float* cnt_ws = sum_ws + (size_t)NNODES * C_OUT;  // NNODES floats

    rse_init_kernel<<<1024, 256, 0, stream>>>(out, sum_ws, cnt_ws);

    int n_groups = (E + 15) / 16;
    int blocks = (n_groups + WAVES_PER_BLOCK - 1) / WAVES_PER_BLOCK;
    if (blocks > 2048) blocks = 2048;
    if (blocks < 1) blocks = 1;
    rse_edge_kernel<<<blocks, 256, 0, stream>>>(G, RSC, src, dst, W, b, out,
                                                sum_ws, cnt_ws, E);

    long nfin = (long)NNODES * C_OUT;
    rse_final_kernel<<<(nfin + 255) / 256, 256, 0, stream>>>(sum_ws, cnt_ws,
                                                             out);
}
